// LabelCrossAttentionDE_28896539968069
// MI455X (gfx1250) — matrix operand order repaired and checked by reading, not compiled, not run on hardware
//
#include <hip/hip_runtime.h>
#include <hip/hip_bf16.h>

typedef __attribute__((ext_vector_type(16))) __bf16 v16bf;
typedef __attribute__((ext_vector_type(8)))  __bf16 v8bf;
typedef __attribute__((ext_vector_type(8)))  float  v8f;

typedef int v4i __attribute__((vector_size(16)));
typedef __attribute__((address_space(1))) v4i g_v4i;   // global 128-bit chunk
typedef __attribute__((address_space(3))) v4i l_v4i;   // LDS 128-bit chunk

#define B_      4
#define S_      4096
#define C_      8192
#define D_      768
#define MASKLEN 2048
#define CT      64    // labels per workgroup
#define ST      32    // S-tile (== WMMA K for GEMM2)
#define DH      384   // D per wave in GEMM2
#define NDT     24    // DH/16 d-subtiles per wave
#define NKT     24    // D_/32 k-steps in GEMM1
#define NEG_INF (-1.0e9f)

// ---- workspace layout (bytes); requires ~63 MB of d_ws
#define WS_QB   0
#define WS_XB   (WS_QB + (size_t)C_ * D_ * 2)
#define WS_XTB  (WS_XB + (size_t)B_ * S_ * D_ * 2)

struct SharedMem {
  alignas(64) __bf16 x [ST][D_];   // row-major x tile (GEMM1 B frags)
  alignas(64) __bf16 xt[D_][ST];   // transposed x tile (GEMM2 B frags)
  alignas(64) __bf16 P [CT][ST];   // softmax probs (GEMM2 A frags)
  float rowmax[2][CT];
  float rowsum[2][CT];
  float Mst[CT];
  float Lst[CT];
  float pSum[2][CT];
  float pSq [2][CT];
  float pOut[2][CT];
};

#if __has_builtin(__builtin_amdgcn_global_load_async_to_lds_b128)
#define HAVE_ASYNC_LDS 1
#else
#define HAVE_ASYNC_LDS 0
#endif

__device__ __forceinline__ void copy16(void* lds_dst, const void* g_src) {
#if HAVE_ASYNC_LDS
  // generic LDS address: low 32 bits are the LDS offset (ISA aperture rules)
  __builtin_amdgcn_global_load_async_to_lds_b128(
      (g_v4i*)(uintptr_t)g_src,
      (l_v4i*)(uint32_t)(uintptr_t)lds_dst,
      0, 0);
#else
  *(uint4*)lds_dst = *(const uint4*)g_src;
#endif
}

__device__ __forceinline__ void wait_async_done() {
#if HAVE_ASYNC_LDS
#if __has_builtin(__builtin_amdgcn_s_wait_asynccnt)
  __builtin_amdgcn_s_wait_asynccnt(0);
#else
  asm volatile("s_wait_asynccnt 0x0" ::: "memory");
#endif
#endif
}

// ---- one-time prep: Q -> bf16, x -> bf16 (row-major) and bf16 transposed per batch
__global__ void prep_kernel(const float* __restrict__ q, const float* __restrict__ x,
                            __bf16* __restrict__ qb, __bf16* __restrict__ xb,
                            __bf16* __restrict__ xtb) {
  const size_t stride = (size_t)gridDim.x * blockDim.x;
  size_t i0 = (size_t)blockIdx.x * blockDim.x + threadIdx.x;
  for (size_t i = i0; i < (size_t)C_ * D_; i += stride) qb[i] = (__bf16)q[i];
  for (size_t i = i0; i < (size_t)B_ * S_ * D_; i += stride) {
    __bf16 e = (__bf16)x[i];
    int    d  = (int)(i % D_);
    int    s  = (int)((i / D_) % S_);
    int    bb = (int)(i / ((size_t)D_ * S_));
    xb[i] = e;
    xtb[((size_t)bb * D_ + d) * S_ + s] = e;
  }
}

__global__ __launch_bounds__(256, 1)
void lca_kernel(const __bf16* __restrict__ xb, const __bf16* __restrict__ xtb,
                const __bf16* __restrict__ qb, const int* __restrict__ mask,
                const float* __restrict__ lnw, const float* __restrict__ lnb,
                const float* __restrict__ ow,  const float* __restrict__ ob,
                float* __restrict__ out) {
  __shared__ SharedMem sm;
  const int tid   = threadIdx.x;
  const int wave  = tid >> 5;
  const int lane  = tid & 31;
  const int lhalf = lane >> 4;   // 0: lanes 0-15, 1: lanes 16-31
  const int ln16  = lane & 15;
  const int cb    = wave & 3;    // c-block (16 rows each)
  const int sw    = wave >> 2;   // s-block for GEMM1 / d-half for GEMM2
  const int b     = blockIdx.x >> 7;            // C_/CT = 128 groups per batch
  const int cbase = (blockIdx.x & 127) * CT;

  for (int i = tid; i < CT; i += 256) { sm.Mst[i] = -3.0e38f; sm.Lst[i] = 0.0f; }

  v8f acc[NDT];
#pragma unroll
  for (int j = 0; j < NDT; ++j) acc[j] = (v8f)(0.0f);

  const __bf16* xrow_g  = xb  + (size_t)b * S_ * D_;   // [S][D] bf16
  const __bf16* xtrow_g = xtb + (size_t)b * D_ * S_;   // [D][S] bf16

  for (int st = 0; st < S_; st += ST) {
    __syncthreads();  // previous iteration's LDS consumers done

    // ---- stage x tile (row-major): contiguous 48 KB copy, 16B granules
    {
      const char* gsrc = (const char*)(xrow_g + (size_t)st * D_);
      char*       ldst = (char*)&sm.x[0][0];
#pragma unroll
      for (int k = tid; k < (ST * D_ * 2) / 16; k += 256)
        copy16(ldst + 16 * k, gsrc + 16 * k);
    }
    // ---- stage x tile (transposed): 768 rows x 64 B from pre-transposed global
    {
#pragma unroll
      for (int k = tid; k < D_ * 4; k += 256) {
        int d = k >> 2, part = k & 3;
        copy16(&sm.xt[d][part * 8], xtrow_g + (size_t)d * S_ + st + part * 8);
      }
    }
    wait_async_done();
    __syncthreads();

    // ---- GEMM1: scores tile [16c x 16s] = Q[16,768] * xT[768,16]
    v8f sc = (v8f)(0.0f);
    {
      const __bf16* qrow = qb + (size_t)(cbase + cb * 16 + ln16) * D_;
      const __bf16* xrow = &sm.x[sw * 16 + ln16][0];
#pragma unroll
      for (int kt = 0; kt < NKT; ++kt) {
        const int d0 = kt * 32;
        union { v16bf v; uint4 q[2]; } A, Bm;
        A.q[0]  = *(const uint4*)(qrow + d0 + 8 * lhalf);        // K 0..7 / 8..15
        A.q[1]  = *(const uint4*)(qrow + d0 + 16 + 8 * lhalf);   // K 16..23 / 24..31
        Bm.q[0] = *(const uint4*)(xrow + d0 + 8 * lhalf);        // K 0..7 / 8..15
        Bm.q[1] = *(const uint4*)(xrow + d0 + 16 + 8 * lhalf);   // K 16..23 / 24..31
        sc = __builtin_amdgcn_wmma_f32_16x16x32_bf16(false, A.v, false, Bm.v,
                                                     (short)0, sc, false, false);
      }
    }

    // ---- mask + online softmax
    const int   sg   = st + sw * 16 + ln16;      // this lane's score column
    const int   mv   = (sg < MASKLEN) ? mask[b * MASKLEN + sg] : 0;
    const float bias = mv ? 0.0f : NEG_INF;

    float s8[8], rm[8];
#pragma unroll
    for (int i = 0; i < 8; ++i) { s8[i] = sc[i] + bias; rm[i] = s8[i]; }
#pragma unroll
    for (int m = 8; m >= 1; m >>= 1)
#pragma unroll
      for (int i = 0; i < 8; ++i) rm[i] = fmaxf(rm[i], __shfl_xor(rm[i], m, 32));
    if (ln16 == 0)
#pragma unroll
      for (int i = 0; i < 8; ++i) sm.rowmax[sw][cb * 16 + i + 8 * lhalf] = rm[i];
    __syncthreads();

    float mnew[8], alpha[8];
#pragma unroll
    for (int i = 0; i < 8; ++i) {
      int   c  = cb * 16 + i + 8 * lhalf;
      float mo = sm.Mst[c];
      float mn = fmaxf(mo, fmaxf(sm.rowmax[0][c], sm.rowmax[1][c]));
      mnew[i] = mn; alpha[i] = __expf(mo - mn);
    }
    float rs[8];
#pragma unroll
    for (int i = 0; i < 8; ++i) { s8[i] = __expf(s8[i] - mnew[i]); rs[i] = s8[i]; }
#pragma unroll
    for (int m = 8; m >= 1; m >>= 1)
#pragma unroll
      for (int i = 0; i < 8; ++i) rs[i] += __shfl_xor(rs[i], m, 32);
    if (ln16 == 0)
#pragma unroll
      for (int i = 0; i < 8; ++i) sm.rowsum[sw][cb * 16 + i + 8 * lhalf] = rs[i];
#pragma unroll
    for (int i = 0; i < 8; ++i)
      sm.P[cb * 16 + i + 8 * lhalf][sw * 16 + ln16] = (__bf16)s8[i];
#pragma unroll
    for (int j = 0; j < NDT; ++j)
#pragma unroll
      for (int i = 0; i < 8; ++i) acc[j][i] *= alpha[i];
    __syncthreads();

    if (sw == 0 && lane < 16) {  // update running softmax state (one wave per c-block)
      int   c  = cb * 16 + lane;
      float mo = sm.Mst[c];
      float mn = fmaxf(mo, fmaxf(sm.rowmax[0][c], sm.rowmax[1][c]));
      sm.Lst[c] = __expf(mo - mn) * sm.Lst[c] + sm.rowsum[0][c] + sm.rowsum[1][c];
      sm.Mst[c] = mn;
    }

    // ---- GEMM2: y[16c x DH] += P[16c x 32s] * x[32s x DH]
    {
      union { v16bf v; uint4 q[2]; } A;
      const __bf16* prow = &sm.P[cb * 16 + ln16][0];
      A.q[0] = *(const uint4*)(prow + 8 * lhalf);
      A.q[1] = *(const uint4*)(prow + 16 + 8 * lhalf);
#pragma unroll
      for (int j = 0; j < NDT; ++j) {
        int d = sw * DH + j * 16 + ln16;            // this lane's output column
        union { v16bf v; uint4 q[2]; } Bm;
        Bm.q[0] = *(const uint4*)(&sm.xt[d][8 * lhalf]);
        Bm.q[1] = *(const uint4*)(&sm.xt[d][16 + 8 * lhalf]);
        acc[j] = __builtin_amdgcn_wmma_f32_16x16x32_bf16(false, A.v, false, Bm.v,
                                                         (short)0, acc[j], false, false);
      }
    }
  }
  __syncthreads();

  // ---- epilogue: softmax denom, LayerNorm stats, fused readout
  float Linv[8], vsum[8], vsq[8];
#pragma unroll
  for (int i = 0; i < 8; ++i) {
    int c = cb * 16 + i + 8 * lhalf;
    Linv[i] = 1.0f / sm.Lst[c];
    vsum[i] = 0.0f; vsq[i] = 0.0f;
  }
#pragma unroll
  for (int j = 0; j < NDT; ++j)
#pragma unroll
    for (int i = 0; i < 8; ++i) {
      float v = acc[j][i] * Linv[i];
      vsum[i] += v; vsq[i] += v * v;
    }
#pragma unroll
  for (int m = 8; m >= 1; m >>= 1)
#pragma unroll
    for (int i = 0; i < 8; ++i) {
      vsum[i] += __shfl_xor(vsum[i], m, 32);
      vsq[i]  += __shfl_xor(vsq[i],  m, 32);
    }
  if (ln16 == 0)
#pragma unroll
    for (int i = 0; i < 8; ++i) {
      sm.pSum[sw][cb * 16 + i + 8 * lhalf] = vsum[i];
      sm.pSq [sw][cb * 16 + i + 8 * lhalf] = vsq[i];
    }
  __syncthreads();

  float mu[8], rstd[8];
#pragma unroll
  for (int i = 0; i < 8; ++i) {
    int   c   = cb * 16 + i + 8 * lhalf;
    float s1  = sm.pSum[0][c] + sm.pSum[1][c];
    float s2  = sm.pSq [0][c] + sm.pSq [1][c];
    float m_  = s1 * (1.0f / (float)D_);
    float var = s2 * (1.0f / (float)D_) - m_ * m_;
    mu[i] = m_; rstd[i] = rsqrtf(var + 1e-5f);
  }
  float o8[8];
#pragma unroll
  for (int i = 0; i < 8; ++i) o8[i] = 0.0f;
#pragma unroll
  for (int j = 0; j < NDT; ++j) {
    int   d  = sw * DH + j * 16 + ln16;
    float lw = lnw[d], lb = lnb[d], w = ow[d];
#pragma unroll
    for (int i = 0; i < 8; ++i) {
      float v = acc[j][i] * Linv[i];
      o8[i] += ((v - mu[i]) * rstd[i] * lw + lb) * w;
    }
  }
#pragma unroll
  for (int m = 8; m >= 1; m >>= 1)
#pragma unroll
    for (int i = 0; i < 8; ++i) o8[i] += __shfl_xor(o8[i], m, 32);
  if (ln16 == 0)
#pragma unroll
    for (int i = 0; i < 8; ++i) sm.pOut[sw][cb * 16 + i + 8 * lhalf] = o8[i];
  __syncthreads();

  if (wave < 4 && lane < 16) {
    int c = wave * 16 + lane;
    out[(size_t)b * C_ + cbase + c] = sm.pOut[0][c] + sm.pOut[1][c] + ob[0];
  }
}

extern "C" void kernel_launch(void* const* d_in, const int* in_sizes, int n_in,
                              void* d_out, int out_size, void* d_ws, size_t ws_size,
                              hipStream_t stream) {
  (void)in_sizes; (void)n_in; (void)out_size; (void)ws_size;
  const float* x    = (const float*)d_in[0];
  const float* q    = (const float*)d_in[1];
  const int*   mask = (const int*)d_in[2];
  const float* lnw  = (const float*)d_in[3];
  const float* lnb  = (const float*)d_in[4];
  const float* ow   = (const float*)d_in[5];
  const float* ob   = (const float*)d_in[6];

  __bf16* qb  = (__bf16*)((char*)d_ws + WS_QB);
  __bf16* xbb = (__bf16*)((char*)d_ws + WS_XB);
  __bf16* xtb = (__bf16*)((char*)d_ws + WS_XTB);

  prep_kernel<<<2048, 256, 0, stream>>>(q, x, qb, xbb, xtb);

  dim3 grid(B_ * (C_ / CT));            // 512 workgroups
  lca_kernel<<<grid, 256, 0, stream>>>(xbb, xtb, qb, mask, lnw, lnb, ow, ob, (float*)d_out);
}
